// GeomLayer_61486751809898
// MI455X (gfx1250) — matrix operand order repaired and checked by reading, not compiled, not run on hardware
//
#include <hip/hip_runtime.h>
#include <hip/hip_bf16.h>

typedef __attribute__((ext_vector_type(16))) _Float16 v16h;
typedef __attribute__((ext_vector_type(8)))  float    v8f;

static constexpr int Bn  = 8;
static constexpr int CIN = 256;
static constexpr int Kn  = 128;
static constexpr int Hn  = 112;
static constexpr int Wn  = 112;
static constexpr int HW  = Hn * Wn;      // 12544
static constexpr int PTILES = HW / 16;   // 784 pixel tiles of 16

// ---------------------------------------------------------------------------
// Pass 1: comb[b,k,p] = sum_c linComb[k,c] * x[b,c,p]  via V_WMMA_F32_16X16X32_F16
// Block = 256 threads = 8 waves. Block -> (b, 16-pixel tile); wave w -> k-tile w.
// All 8 waves read the same x tile (WGP$ reuse); linComb (128KB) lives in L2.
// ---------------------------------------------------------------------------
__global__ __launch_bounds__(256) void geom_gemm_wmma(
    const float* __restrict__ x,        // [B, CIN, HW]
    const float* __restrict__ linComb,  // [K, CIN]
    float* __restrict__ comb)           // [B, K, HW]
{
  const int wave  = threadIdx.x >> 5;
  const int lane  = threadIdx.x & 31;
  const int row   = lane & 15;          // M for A rows, N (pixel) for B/D cols
  const int khalf = (lane >> 4) & 1;

  const int b    = blockIdx.x / PTILES;
  const int pt   = blockIdx.x % PTILES;
  const int pix0 = pt * 16;
  const int k0   = wave * 16;

  const float* __restrict__ xb = x + (size_t)b * CIN * HW;

  v8f acc = {};
#pragma unroll
  for (int c0 = 0; c0 < CIN; c0 += 32) {
    // A: linComb rows (16 k x 32 c), ISA 16-bit A layout
    v16h a;
    const float* lrow = linComb + (size_t)(k0 + row) * CIN + c0;
#pragma unroll
    for (int i = 0; i < 8; ++i) {
      a[i]     = (_Float16)lrow[khalf * 8 + i];
      a[8 + i] = (_Float16)lrow[16 + khalf * 8 + i];
    }
    // B: x tile (32 c x 16 px), ISA 16-bit B layout: lane<16 -> c0..c0+7, c0+16..c0+23,
    // lane>=16 -> c0+8..c0+15, c0+24..c0+31, column = pix0 + (lane&15)
    v16h bm;
    const float* xcol = xb + (size_t)(c0 + khalf * 8) * HW + pix0 + row;
#pragma unroll
    for (int i = 0; i < 16; ++i) {
      bm[i] = (_Float16)xcol[(size_t)(i < 8 ? i : i + 8) * HW];
    }
    acc = __builtin_amdgcn_wmma_f32_16x16x32_f16(
        /*neg_a=*/false, a, /*neg_b=*/false, bm,
        /*c_mod=*/(short)0, acc, /*reuse_a=*/false, /*reuse_b=*/false);
  }

  // D layout: lane<16 -> M=r, N=lane ; lane>=16 -> M=r+8, N=lane-16
  float* __restrict__ out =
      comb + ((size_t)b * Kn + k0 + khalf * 8) * HW + pix0 + row;
#pragma unroll
  for (int r = 0; r < 8; ++r) {
    out[(size_t)r * HW] = acc[r];
  }
}

// ---------------------------------------------------------------------------
// Pass 2: per output element, bilinear gather from comb (geoParams) and the
// analytic ones-warp (boxParams: sum of tap-weight * in-bounds, no gather).
// d_out = [finalPred | warped | box], each B*K*HW floats.
// ---------------------------------------------------------------------------
__device__ __forceinline__ float in_bounds(float xc, float yc) {
  return (xc >= 0.0f && xc <= (float)(Wn - 1) &&
          yc >= 0.0f && yc <= (float)(Hn - 1)) ? 1.0f : 0.0f;
}

__global__ __launch_bounds__(256) void geom_warp(
    const float* __restrict__ comb,   // [B, K, HW]
    const float* __restrict__ geo,    // [K, 2, 3]
    const float* __restrict__ boxp,   // [K, 2, 3]
    float* __restrict__ out)          // [3, B, K, HW]
{
  const int idx = blockIdx.x * 256 + threadIdx.x;   // < B*K*HW
  const int pix = idx % HW;
  const int bk  = idx / HW;                          // uniform per block (256 | HW)
  const int k   = bk % Kn;
  const int w   = pix % Wn;
  const int h   = pix / Wn;

  const float xs = (2.0f * (float)w + 1.0f) * (1.0f / (float)Wn) - 1.0f;
  const float ys = (2.0f * (float)h + 1.0f) * (1.0f / (float)Hn) - 1.0f;

  // ---- geometric warp: 4-tap bilinear gather from comb[b,k] ----
  const float* g = geo + (size_t)k * 6;
  const float gx = g[0] * xs + g[1] * ys + g[2];
  const float gy = g[3] * xs + g[4] * ys + g[5];
  const float ix = (gx + 1.0f) * ((float)Wn * 0.5f) - 0.5f;
  const float iy = (gy + 1.0f) * ((float)Hn * 0.5f) - 0.5f;
  const float x0 = floorf(ix), y0 = floorf(iy);
  const float wx1 = ix - x0, wx0 = 1.0f - wx1;
  const float wy1 = iy - y0, wy0 = 1.0f - wy1;

  const float* __restrict__ src = comb + (size_t)bk * HW;
  float warped = 0.0f;
#pragma unroll
  for (int t = 0; t < 4; ++t) {
    const float xc = x0 + (float)(t & 1);
    const float yc = y0 + (float)(t >> 1);
    const float wgt = ((t & 1) ? wx1 : wx0) * ((t >> 1) ? wy1 : wy0);
    const float m = in_bounds(xc, yc);
    const int xi = (int)fminf(fmaxf(xc, 0.0f), (float)(Wn - 1));
    const int yi = (int)fminf(fmaxf(yc, 0.0f), (float)(Hn - 1));
    warped += src[yi * Wn + xi] * (wgt * m);
  }

  // ---- box warp of all-ones image: analytic, no gather ----
  const float* bp = boxp + (size_t)k * 6;
  const float hx = bp[0] * xs + bp[1] * ys + bp[2];
  const float hy = bp[3] * xs + bp[4] * ys + bp[5];
  const float jx = (hx + 1.0f) * ((float)Wn * 0.5f) - 0.5f;
  const float jy = (hy + 1.0f) * ((float)Hn * 0.5f) - 0.5f;
  const float bx0 = floorf(jx), by0 = floorf(jy);
  const float ux1 = jx - bx0, ux0 = 1.0f - ux1;
  const float uy1 = jy - by0, uy0 = 1.0f - uy1;
  const float boxv = ux0 * uy0 * in_bounds(bx0,        by0)
                   + ux1 * uy0 * in_bounds(bx0 + 1.0f, by0)
                   + ux0 * uy1 * in_bounds(bx0,        by0 + 1.0f)
                   + ux1 * uy1 * in_bounds(bx0 + 1.0f, by0 + 1.0f);

  const size_t N = (size_t)Bn * Kn * HW;
  out[idx]         = warped * boxv;  // finalPred
  out[N + idx]     = warped;         // warped
  out[2 * N + idx] = boxv;           // box
}

// ---------------------------------------------------------------------------
extern "C" void kernel_launch(void* const* d_in, const int* in_sizes, int n_in,
                              void* d_out, int out_size, void* d_ws, size_t ws_size,
                              hipStream_t stream) {
  (void)in_sizes; (void)n_in; (void)out_size; (void)ws_size;
  const float* x    = (const float*)d_in[0];  // [B, CIN, H, W]
  const float* lin  = (const float*)d_in[1];  // [K, CIN]
  const float* geo  = (const float*)d_in[2];  // [K, 2, 3]
  const float* boxp = (const float*)d_in[3];  // [K, 2, 3]
  float* out  = (float*)d_out;                // [finalPred|warped|box]
  float* comb = (float*)d_ws;                 // needs B*K*HW*4 = 51.4 MB scratch

  geom_gemm_wmma<<<Bn * PTILES, 256, 0, stream>>>(x, lin, comb);

  const int total = Bn * Kn * HW;             // 12,845,056 (divisible by 256)
  geom_warp<<<total / 256, 256, 0, stream>>>(comb, geo, boxp, out);
}
